// WTM_A_ML_16879221473450
// MI455X (gfx1250) — compile-verified, matrix operand order checked by reading, not run on hardware
//
#include <hip/hip_runtime.h>
#include <cstdint>
#include <cstddef>

// ---------------------------------------------------------------------------
// Types
// ---------------------------------------------------------------------------
typedef __bf16 bf16_t;
typedef __attribute__((ext_vector_type(16))) __bf16        v16bf;
typedef __attribute__((ext_vector_type(8)))  float         v8f;
typedef __attribute__((ext_vector_type(4)))  unsigned int  uint4v;

#define DEV __device__ __forceinline__

DEV bf16_t to_bf(float f) {
  unsigned u = __float_as_uint(f);
  unsigned r = (u + 0x7FFFu + ((u >> 16) & 1u)) >> 16;  // round-to-nearest-even
  unsigned short us = (unsigned short)r;
  return __builtin_bit_cast(bf16_t, us);
}

DEV float wave_sum32(float v) {
#pragma unroll
  for (int off = 16; off > 0; off >>= 1) v += __shfl_xor(v, off, 32);
  return v;
}

DEV v8f wmma_bf16(v16bf a, v16bf b, v8f c) {
#if defined(__gfx1250__)
  return __builtin_amdgcn_wmma_f32_16x16x32_bf16(false, a, false, b, (short)0, c,
                                                 false, false);
#else
  return c;
#endif
}

// ---------------------------------------------------------------------------
// Fragment loaders (16-bit A 16x32 / B 32x16 layout, ISA 7.12.2):
// lane L: row/col = L%16, K-base = (L/16)*8; VGPRs 0-3 hold K..K+7,
// VGPRs 4-7 hold K+16..K+23  -> two contiguous 16B chunks per lane.
// ---------------------------------------------------------------------------
DEV v16bf load_frag_dense(const bf16_t* base, int ld, int lane) {
  int r  = lane & 15;
  int kb = (lane >> 4) << 3;
  const bf16_t* p = base + (size_t)r * ld + kb;
  union { uint4v u[2]; v16bf v; } f;
  f.u[0] = *reinterpret_cast<const uint4v*>(p);
  f.u[1] = *reinterpret_cast<const uint4v*>(p + 16);
  return f.v;
}

// implicit-GEMM gather for 3x3 conv, C=256, K ordered tap-major: k = tap*256 + c
DEV uint4v conv_chunk(const bf16_t* in, int b, int y, int x, int H, int W, int kg) {
  int tap = kg >> 8;
  int c   = kg & 255;
  int ny  = y + (tap / 3) - 1;
  int nx  = x + (tap % 3) - 1;
  if (ny < 0 || ny >= H || nx < 0 || nx >= W) {
    uint4v z = {0u, 0u, 0u, 0u};
    return z;
  }
  return *reinterpret_cast<const uint4v*>(in + (((size_t)(b * H + ny) * W) + nx) * 256 + c);
}

DEV v16bf load_frag_conv(const bf16_t* in, int b, int y, int x, int kk, int lane,
                         int H, int W) {
  int kb = kk + ((lane >> 4) << 3);
  union { uint4v u[2]; v16bf v; } f;
  f.u[0] = conv_chunk(in, b, y, x, H, W, kb);
  f.u[1] = conv_chunk(in, b, y, x, H, W, kb + 16);
  return f.v;
}

// ---------------------------------------------------------------------------
// GEMM: D(MxN) = A(MxK) * W(NxK)^T, bf16 in, f32 accumulate, fused epilogues.
// One wave = 16 x (16*NT) tile, software pipelined over K with two fragment
// buffers alternating roles (K-loop manually unrolled by 2 -> no rotate moves,
// partial s_wait_loadcnt before each WMMA chain).
// Requires: M%16==0, K%64==0, N%(16*NT)==0.
// EPI: 0 bias->f32 | 1 res+gamma*(acc+bias)->f32 | 2 gelu->bf16 |
//      3 BN->f32   | 4 BN+relu->bf16             | 5 BN+relu->NCHW f32 (col<NOUT)
// ---------------------------------------------------------------------------
template <int EPI, bool ACONV, int NT>
__global__ __launch_bounds__(256) void gemm_bf16_kernel(
    const bf16_t* __restrict__ A, const bf16_t* __restrict__ Wt,
    const float* __restrict__ bias, const float* p1, const float* __restrict__ p2,
    const float* __restrict__ p3, const float* __restrict__ p4,
    float* outF, bf16_t* __restrict__ outB,
    int M, int N, int Kd, int H, int W, int NOUT) {
  const int lane   = threadIdx.x & 31;
  const int wav    = blockIdx.x * (blockDim.x >> 5) + (threadIdx.x >> 5);
  const int tilesM = M >> 4;
  const int tilesN = N / (16 * NT);
  if (wav >= tilesM * tilesN) return;  // wave-uniform: EXEC stays all-ones
  const int tm    = wav % tilesM;
  const int tn    = wav / tilesM;
  const int m0    = tm << 4;
  const int nbase = tn * (16 * NT);

  // pixel decode for the conv gather (depends only on this lane's A-row)
  int cb = 0, cy = 0, cx = 0;
  if (ACONV) {
    int m = m0 + (lane & 15);
    cx = m % W;
    cy = (m / W) % H;
    cb = m / (W * H);
  }

  v8f acc[NT];
#pragma unroll
  for (int s = 0; s < NT; ++s) acc[s] = v8f{};

  const bf16_t* wb = Wt + (size_t)nbase * Kd;

  auto loadA = [&](int kk) -> v16bf {
    if (ACONV) return load_frag_conv(A, cb, cy, cx, kk, lane, H, W);
    return load_frag_dense(A + (size_t)m0 * Kd + kk, Kd, lane);
  };
  auto loadB = [&](int kk, v16bf* dst) {
#pragma unroll
    for (int s = 0; s < NT; ++s)
      dst[s] = load_frag_dense(wb + (size_t)(16 * s) * Kd + kk, Kd, lane);
  };
  auto mmchain = [&](v16bf a, const v16bf* b) {
#pragma unroll
    for (int s = 0; s < NT; ++s) acc[s] = wmma_bf16(a, b[s], acc[s]);
  };

  // two-deep pipeline, buffers alternate roles (K/32 is even for all our GEMMs)
  v16bf a0, a1;
  v16bf b0[NT], b1[NT];
  a0 = loadA(0);
  loadB(0, b0);
  int kk = 32;
  for (; kk + 32 < Kd; kk += 64) {
    a1 = loadA(kk);
    loadB(kk, b1);
    mmchain(a0, b0);
    a0 = loadA(kk + 32);
    loadB(kk + 32, b0);
    mmchain(a1, b1);
  }
  a1 = loadA(Kd - 32);
  loadB(Kd - 32, b1);
  mmchain(a0, b0);
  mmchain(a1, b1);

  // C/D layout: VGPR r, lanes 0-15 -> row m0+r, lanes 16-31 -> row m0+r+8; col = lane%16
  const int rofs = (lane >> 4) << 3;
  const int cl   = lane & 15;
#pragma unroll
  for (int s = 0; s < NT; ++s) {
    const int col = nbase + (s << 4) + cl;
#pragma unroll
    for (int r = 0; r < 8; ++r) {
      const int row = m0 + r + rofs;
      float v = acc[s][r];
      if (EPI == 0) {
        outF[(size_t)row * N + col] = v + bias[col];
      } else if (EPI == 1) {
        outF[(size_t)row * N + col] =
            p1[(size_t)row * N + col] + p2[col] * (v + bias[col]);
      } else if (EPI == 2) {
        float t = v + bias[col];
        outB[(size_t)row * N + col] =
            to_bf(0.5f * t * (1.0f + erff(t * 0.70710678118654752f)));
      } else if (EPI == 3) {
        float t = v + bias[col];
        outF[(size_t)row * N + col] =
            (t - p1[col]) * (p2[col] * rsqrtf(p4[col] + 1e-5f)) + p3[col];
      } else if (EPI == 4) {
        float t = v + bias[col];
        float y = (t - p1[col]) * (p2[col] * rsqrtf(p4[col] + 1e-5f)) + p3[col];
        outB[(size_t)row * N + col] = to_bf(fmaxf(y, 0.0f));
      } else {  // EPI == 5
        if (col < NOUT) {
          float t = v + bias[col];
          float y = (t - p1[col]) * (p2[col] * rsqrtf(p4[col] + 1e-5f)) + p3[col];
          y = fmaxf(y, 0.0f);
          int xx = row % W;
          int yy = (row / W) % H;
          int bb = row / (W * H);
          outF[(((size_t)(bb * NOUT + col)) * H + yy) * W + xx] = y;
        }
      }
    }
  }
}

// ---------------------------------------------------------------------------
// Neighborhood window / bias-start helpers (K=5, NS=2)
// ---------------------------------------------------------------------------
DEV int win_start(int i, int L, int d) {
  if (d <= 1) {
    int s = i - 2;
    if (s < 0) s = 0;
    if (i + 2 >= L) s += L - i - 3;
    return s;
  }
  int ni = i - 2 * d;
  if (ni < 0) return i % d;
  if (i + 2 * d >= L) {
    int im = i % d;
    int a  = (L / d) * d;
    int bb = L - a;
    if (im < bb) return L - bb + im - 4 * d;
    return a + im - 5 * d;
  }
  return ni;
}
DEV int pb_start(int i, int L, int d) {
  if (d <= 1) {
    int s = 2;
    if (i < 2) s += 2 - i;
    if (i + 2 >= L) s += L - i - 3;
    return s;
  }
  if (i - 2 * d < 0) return 4 - i / d;
  if (i + 2 * d >= L) return (L - i - 1) / d;
  return 2;
}

// ---------------------------------------------------------------------------
// Neighborhood attention: one wave per (pixel, head); lane = channel (hd=32)
// qkv: [M, 768] f32 (q|k|v, each heads*32), already biased. q scaled here.
// ---------------------------------------------------------------------------
__global__ __launch_bounds__(256) void nat_attn_kernel(
    const float* __restrict__ qkv, const float* __restrict__ rpb,
    bf16_t* __restrict__ out, int M, int H, int W, int d) {
  int lane = threadIdx.x & 31;
  int wid  = blockIdx.x * 8 + (threadIdx.x >> 5);
  int head = wid & 7;
  int m    = wid >> 3;
  if (m >= M) return;
  int j = m % W;
  int i = (m / W) % H;
  int b = m / (H * W);
  const float scale = 0.17677669529663687f;  // 32^-0.5
  float qc = qkv[(size_t)m * 768 + head * 32 + lane] * scale;
  int si = win_start(i, H, d), sj = win_start(j, W, d);
  int bi = pb_start(i, H, d), bj = pb_start(j, W, d);

  float lg[25];
#pragma unroll
  for (int ky = 0; ky < 5; ++ky) {
    int nmr = (b * H + si + ky * d) * W + sj;
#pragma unroll
    for (int kx = 0; kx < 5; ++kx) {
      int nm = nmr + kx * d;
      float p = qc * qkv[(size_t)nm * 768 + 256 + head * 32 + lane];
      p = wave_sum32(p);
      lg[ky * 5 + kx] = p + rpb[head * 81 + (bi + ky) * 9 + (bj + kx)];
    }
  }
  float mx = lg[0];
#pragma unroll
  for (int p = 1; p < 25; ++p) mx = fmaxf(mx, lg[p]);
  float ssum = 0.0f;
#pragma unroll
  for (int p = 0; p < 25; ++p) {
    lg[p] = __expf(lg[p] - mx);
    ssum += lg[p];
  }
  float inv = 1.0f / ssum;
  float o = 0.0f;
#pragma unroll
  for (int ky = 0; ky < 5; ++ky) {
    int nmr = (b * H + si + ky * d) * W + sj;
#pragma unroll
    for (int kx = 0; kx < 5; ++kx) {
      int nm = nmr + kx * d;
      o += lg[ky * 5 + kx] * qkv[(size_t)nm * 768 + 512 + head * 32 + lane];
    }
  }
  out[(size_t)m * 256 + head * 32 + lane] = to_bf(o * inv);
}

// ---------------------------------------------------------------------------
// LayerNorm over C=256: one wave per pixel, lane handles 8 channels. -> bf16
// ---------------------------------------------------------------------------
__global__ __launch_bounds__(256) void ln_kernel(
    const float* __restrict__ x, const float* __restrict__ w,
    const float* __restrict__ b, bf16_t* __restrict__ out, int M) {
  int lane = threadIdx.x & 31;
  int m    = blockIdx.x * 8 + (threadIdx.x >> 5);
  if (m >= M) return;
  const float* xp = x + (size_t)m * 256;
  float vals[8], s = 0.0f, s2 = 0.0f;
#pragma unroll
  for (int j = 0; j < 8; ++j) {
    float v = xp[lane + 32 * j];
    vals[j] = v;
    s += v;
    s2 += v * v;
  }
  s  = wave_sum32(s);
  s2 = wave_sum32(s2);
  float mean = s * (1.0f / 256.0f);
  float var  = s2 * (1.0f / 256.0f) - mean * mean;
  float rstd = rsqrtf(var + 1e-5f);
  bf16_t* op = out + (size_t)m * 256;
#pragma unroll
  for (int j = 0; j < 8; ++j) {
    int c = lane + 32 * j;
    op[c] = to_bf((vals[j] - mean) * rstd * w[c] + b[c]);
  }
}

// ---------------------------------------------------------------------------
// Elementwise helpers
// ---------------------------------------------------------------------------
__global__ void cvt_f32_bf16_kernel(const float* __restrict__ in,
                                    bf16_t* __restrict__ out, long n) {
  long i = (long)blockIdx.x * blockDim.x + threadIdx.x;
  if (i < n) out[i] = to_bf(in[i]);
}

__global__ void nchw_to_nhwc_kernel(const float* __restrict__ in,
                                    float* __restrict__ out, int Bn, int C,
                                    int H, int W) {
  long i = (long)blockIdx.x * blockDim.x + threadIdx.x;
  long total = (long)Bn * C * H * W;
  if (i >= total) return;
  int x = (int)(i % W);
  long t = i / W;
  int y = (int)(t % H);
  t /= H;
  int c = (int)(t % C);
  int b = (int)(t / C);
  out[(((long)(b * H + y) * W + x) * C) + c] = in[i];
}

// bilinear upsample, NHWC, align-corners weights (linspace 0..h-1 over H pts)
__global__ void upsample_kernel(const float* __restrict__ in,
                                float* __restrict__ out, int Bn, int h, int w,
                                int H, int W) {
  long i = (long)blockIdx.x * blockDim.x + threadIdx.x;
  long total = (long)Bn * H * W * 256;
  if (i >= total) return;
  int c = (int)(i & 255);
  long t = i >> 8;
  int X = (int)(t % W);
  int Y = (int)((t / W) % H);
  int b = (int)(t / ((long)W * H));
  float fy = (float)Y * (float)(h - 1) / (float)(H - 1);
  float fx = (float)X * (float)(w - 1) / (float)(W - 1);
  int y0 = (int)floorf(fy);
  int x0 = (int)floorf(fx);
  int y1 = y0 + 1 < h ? y0 + 1 : h - 1;
  int x1 = x0 + 1 < w ? x0 + 1 : w - 1;
  float wy = fy - (float)y0, wx = fx - (float)x0;
  const float* base = in + (long)b * h * w * 256 + c;
  float a = base[((long)y0 * w + x0) * 256];
  float bb = base[((long)y0 * w + x1) * 256];
  float cc = base[((long)y1 * w + x0) * 256];
  float e = base[((long)y1 * w + x1) * 256];
  out[i] = a * (1.f - wy) * (1.f - wx) + bb * (1.f - wy) * wx +
           cc * wy * (1.f - wx) + e * wy * wx;
}

// cat([feat(NCHW), up(NHWC)], C) -> bf16 [pixels, 512]
__global__ void cat_kernel(const float* __restrict__ feat,
                           const float* __restrict__ up,
                           bf16_t* __restrict__ out, int Bn, int H, int W) {
  long i = (long)blockIdx.x * blockDim.x + threadIdx.x;
  long total = (long)Bn * H * W * 512;
  if (i >= total) return;
  int c = (int)(i & 511);
  long t = i >> 9;
  int x = (int)(t % W);
  int y = (int)((t / W) % H);
  int b = (int)(t / ((long)W * H));
  float v = (c < 256) ? feat[((long)(b * 256 + c) * H + y) * W + x]
                      : up[t * 256 + (c - 256)];
  out[i] = to_bf(v);
}

// (O,C,3,3) f32 -> bf16 [Opad, 2304], k = tap*256 + c, zero-fill o>=Oreal
__global__ void pack_conv_w_kernel(const float* __restrict__ w,
                                   bf16_t* __restrict__ out, int Oreal,
                                   int Opad) {
  long i = (long)blockIdx.x * blockDim.x + threadIdx.x;
  long total = (long)Opad * 2304;
  if (i >= total) return;
  int k   = (int)(i % 2304);
  int o   = (int)(i / 2304);
  int tap = k >> 8;
  int c   = k & 255;
  float v = (o < Oreal) ? w[(long)o * 2304 + (long)c * 9 + tap] : 0.0f;
  out[i] = to_bf(v);
}

// ---------------------------------------------------------------------------
// Host side
// ---------------------------------------------------------------------------
static inline int cdiv(long a, long b) { return (int)((a + b - 1) / b); }

template <int EPI, bool ACONV, int NT>
static void gemm_launch(hipStream_t stream, const bf16_t* A, const bf16_t* Wt,
                        const float* bias, const float* p1, const float* p2,
                        const float* p3, const float* p4, float* outF,
                        bf16_t* outB, int M, int N, int Kd, int H = 0,
                        int W = 0, int NOUT = 0) {
  int waves = (M >> 4) * (N / (16 * NT));
  dim3 grid((waves + 7) / 8);
  gemm_bf16_kernel<EPI, ACONV, NT><<<grid, dim3(256), 0, stream>>>(
      A, Wt, bias, p1, p2, p3, p4, outF, outB, M, N, Kd, H, W, NOUT);
}

static void cvt_bf16(hipStream_t s, const float* in, bf16_t* out, long n) {
  cvt_f32_bf16_kernel<<<cdiv(n, 256), 256, 0, s>>>(in, out, n);
}

extern "C" void kernel_launch(void* const* d_in, const int* in_sizes, int n_in,
                              void* d_out, int out_size, void* d_ws,
                              size_t ws_size, hipStream_t stream) {
  (void)in_sizes; (void)n_in; (void)out_size; (void)ws_size;
  const float* x0     = (const float*)d_in[0];
  const float* x1     = (const float*)d_in[1];
  const float* x2     = (const float*)d_in[2];
  const float* x3     = (const float*)d_in[3];
  const float* ln1_w  = (const float*)d_in[4];
  const float* ln1_b  = (const float*)d_in[5];
  const float* qkv_w  = (const float*)d_in[6];
  const float* qkv_b  = (const float*)d_in[7];
  const float* rpb    = (const float*)d_in[8];
  const float* proj_w = (const float*)d_in[9];
  const float* proj_b = (const float*)d_in[10];
  const float* ln2_w  = (const float*)d_in[11];
  const float* ln2_b  = (const float*)d_in[12];
  const float* fc1_w  = (const float*)d_in[13];
  const float* fc1_b  = (const float*)d_in[14];
  const float* fc2_w  = (const float*)d_in[15];
  const float* fc2_b  = (const float*)d_in[16];
  const float* g1     = (const float*)d_in[17];
  const float* g2     = (const float*)d_in[18];
  const float* fuse_w = (const float*)d_in[19];
  const float* fuse_b = (const float*)d_in[20];
  const float* fbn_w  = (const float*)d_in[21];
  const float* fbn_b  = (const float*)d_in[22];
  const float* fbn_m  = (const float*)d_in[23];
  const float* fbn_v  = (const float*)d_in[24];
  const float* lc1_w  = (const float*)d_in[25];
  const float* lc1_b  = (const float*)d_in[26];
  const float* lbn1_w = (const float*)d_in[27];
  const float* lbn1_b = (const float*)d_in[28];
  const float* lbn1_m = (const float*)d_in[29];
  const float* lbn1_v = (const float*)d_in[30];
  const float* lc2_w  = (const float*)d_in[31];
  const float* lc2_b  = (const float*)d_in[32];
  const float* lbn2_w = (const float*)d_in[33];
  const float* lbn2_b = (const float*)d_in[34];
  const float* lbn2_m = (const float*)d_in[35];
  const float* lbn2_v = (const float*)d_in[36];

  const long MAXPIX = 4L * 64 * 48;  // 12288
  char* ws = (char*)d_ws;
  size_t off = 0;
  auto alloc = [&](size_t bytes) -> char* {
    char* p = ws + off;
    off += (bytes + 255) & ~(size_t)255;
    return p;
  };
  bf16_t* WQKV  = (bf16_t*)alloc(8ull * 768 * 256 * 2);
  bf16_t* WPROJ = (bf16_t*)alloc(8ull * 256 * 256 * 2);
  bf16_t* WFC1  = (bf16_t*)alloc(8ull * 1024 * 256 * 2);
  bf16_t* WFC2  = (bf16_t*)alloc(8ull * 256 * 1024 * 2);
  bf16_t* WFUSE = (bf16_t*)alloc(3ull * 256 * 512 * 2);
  bf16_t* WL1   = (bf16_t*)alloc(256ull * 2304 * 2);
  bf16_t* WL2   = (bf16_t*)alloc(32ull * 2304 * 2);
  float*  XF    = (float*)alloc((size_t)MAXPIX * 256 * 4);
  bf16_t* LNB   = (bf16_t*)alloc((size_t)MAXPIX * 256 * 2);
  float*  QKVF  = (float*)alloc((size_t)MAXPIX * 768 * 4);
  bf16_t* ATTNB = (bf16_t*)alloc((size_t)MAXPIX * 256 * 2);
  bf16_t* FC1B  = (bf16_t*)alloc((size_t)MAXPIX * 1024 * 2);
  // reuse (disjoint in time): upsample/cat live in the qkv arena;
  // conv input reuses LNB, conv1 output reuses ATTNB.
  float*  UPF   = QKVF;
  bf16_t* CATB  = (bf16_t*)((char*)QKVF + (size_t)MAXPIX * 256 * 4);
  bf16_t* CIN   = LNB;
  bf16_t* C1B   = ATTNB;

  // ---- weight conversions (fp32 -> bf16), once per call ----
  cvt_bf16(stream, qkv_w,  WQKV,  8L * 768 * 256);
  cvt_bf16(stream, proj_w, WPROJ, 8L * 256 * 256);
  cvt_bf16(stream, fc1_w,  WFC1,  8L * 1024 * 256);
  cvt_bf16(stream, fc2_w,  WFC2,  8L * 256 * 1024);
  cvt_bf16(stream, fuse_w, WFUSE, 3L * 256 * 512);
  pack_conv_w_kernel<<<cdiv(256L * 2304, 256), 256, 0, stream>>>(lc1_w, WL1, 256, 256);
  pack_conv_w_kernel<<<cdiv(32L * 2304, 256), 256, 0, stream>>>(lc2_w, WL2, 17, 32);

  auto run_layer = [&](int l, int H, int W, int d) {
    int M = 4 * H * W;
    // LN1 -> bf16
    ln_kernel<<<cdiv(M, 8), 256, 0, stream>>>(XF, ln1_w + l * 256, ln1_b + l * 256, LNB, M);
    // qkv = h @ qkv_w.T + qkv_b
    gemm_launch<0, false, 4>(stream, LNB, WQKV + (size_t)l * 768 * 256, qkv_b + l * 768,
                             nullptr, nullptr, nullptr, nullptr, QKVF, nullptr, M, 768, 256);
    // neighborhood attention -> bf16
    nat_attn_kernel<<<M, 256, 0, stream>>>(QKVF, rpb + (size_t)l * 648, ATTNB, M, H, W, d);
    // x = x + g1 * (o @ proj_w.T + proj_b)   (in-place on XF)
    gemm_launch<1, false, 4>(stream, ATTNB, WPROJ + (size_t)l * 256 * 256, proj_b + l * 256,
                             XF, g1 + l * 256, nullptr, nullptr, XF, nullptr, M, 256, 256);
    // LN2 -> bf16
    ln_kernel<<<cdiv(M, 8), 256, 0, stream>>>(XF, ln2_w + l * 256, ln2_b + l * 256, LNB, M);
    // fc1 + exact GELU -> bf16
    gemm_launch<2, false, 4>(stream, LNB, WFC1 + (size_t)l * 1024 * 256, fc1_b + l * 1024,
                             nullptr, nullptr, nullptr, nullptr, nullptr, FC1B, M, 1024, 256);
    // x = x + g2 * (gelu @ fc2_w.T + fc2_b)
    gemm_launch<1, false, 4>(stream, FC1B, WFC2 + (size_t)l * 256 * 1024, fc2_b + l * 256,
                             XF, g2 + l * 256, nullptr, nullptr, XF, nullptr, M, 256, 1024);
  };

  auto stage_up_fuse = [&](const float* feat, int i, int h, int w, int H, int W) {
    int M = 4 * H * W;
    upsample_kernel<<<cdiv((long)M * 256, 256), 256, 0, stream>>>(XF, UPF, 4, h, w, H, W);
    cat_kernel<<<cdiv((long)M * 512, 256), 256, 0, stream>>>(feat, UPF, CATB, 4, H, W);
    gemm_launch<3, false, 4>(stream, CATB, WFUSE + (size_t)i * 256 * 512, fuse_b + i * 256,
                             fbn_m + i * 256, fbn_w + i * 256, fbn_b + i * 256,
                             fbn_v + i * 256, XF, nullptr, M, 256, 512);
  };

  // ---- stage 0: 8x6 ----
  nchw_to_nhwc_kernel<<<cdiv(4L * 256 * 8 * 6, 256), 256, 0, stream>>>(x3, XF, 4, 256, 8, 6);
  run_layer(0, 8, 6, 1);
  run_layer(1, 8, 6, 1);
  // ---- stage 1: 16x12 ----
  stage_up_fuse(x2, 0, 8, 6, 16, 12);
  run_layer(2, 16, 12, 2);
  run_layer(3, 16, 12, 2);
  // ---- stage 2: 32x24 ----
  stage_up_fuse(x1, 1, 16, 12, 32, 24);
  run_layer(4, 32, 24, 4);
  run_layer(5, 32, 24, 4);
  // ---- stage 3: 64x48 ----
  stage_up_fuse(x0, 2, 32, 24, 64, 48);
  run_layer(6, 64, 48, 8);
  run_layer(7, 64, 48, 8);

  // ---- head: conv3x3 + BN + ReLU (implicit GEMM, K = 9*256), then conv2 -> d_out NCHW
  int M = 4 * 64 * 48;
  cvt_bf16(stream, XF, CIN, (long)M * 256);
  gemm_launch<4, true, 4>(stream, CIN, WL1, lc1_b, lbn1_m, lbn1_w, lbn1_b, lbn1_v,
                          nullptr, C1B, M, 256, 2304, 64, 48);
  gemm_launch<5, true, 2>(stream, C1B, WL2, lc2_b, lbn2_m, lbn2_w, lbn2_b, lbn2_v,
                          (float*)d_out, nullptr, M, 32, 2304, 64, 48, 17);
}
